// PaperModel_55181739819295
// MI455X (gfx1250) — matrix operand order repaired and checked by reading, not compiled, not run on hardware
//
#include <hip/hip_runtime.h>

typedef __attribute__((ext_vector_type(16))) __bf16 v16bf;
typedef __attribute__((ext_vector_type(8)))  float  v8f;

#define N_PAPERS 15000
#define SEQ_LEN  80
#define EMBED    128
#define NFILT    64
#define TXT_DIM  256
#define MID      256
#define OUT_DIM  128
#define N_EDGES  480000
#define N_BRIDGE 1024
#define N_SEL    4096

// ---------------------------------------------------------------- helpers
__device__ __forceinline__ unsigned short f2bf(float f) {
  union { float f; unsigned u; } x; x.f = f;
  unsigned r = x.u + 0x7FFFu + ((x.u >> 16) & 1u);   // round-to-nearest-even
  return (unsigned short)(r >> 16);
}

// f32 -> bf16 bulk convert (embed table only)
__global__ void cvt_bf16_kernel(const float* __restrict__ in,
                                unsigned short* __restrict__ out, int n) {
  for (int i = blockIdx.x * blockDim.x + threadIdx.x; i < n;
       i += gridDim.x * blockDim.x)
    out[i] = f2bf(in[i]);
}

// Pack B[K,N] (f32, row-major) into WMMA-fragment-major bf16:
//   frag index = (k/32)*(N/16) + n/16
//   lane       = (((k%32)/8)%2)*16 + n%16     (B layout: lane = N col, K group)
//   elem e     = k%8 + ((k%32)/16)*8          (table order: K = 8g+e / 16+8g+e-8)
// Each lane's 16 elems land contiguous -> fragment load = 2x b128.
__global__ void pack_b_kernel(const float* __restrict__ B,
                              unsigned short* __restrict__ out, int K, int N) {
  int total = K * N;
  int ntiles = N >> 4;
  for (int i = blockIdx.x * blockDim.x + threadIdx.x; i < total;
       i += gridDim.x * blockDim.x) {
    int k = i / N, n = i - k * N;
    int kt = k >> 5, kr = k & 31;
    int g = (kr >> 3) & 1, e = (kr & 7) | ((kr >> 4) << 3);
    int nt = n >> 4, nl = n & 15;
    int lane = g * 16 + nl;
    out[(((kt * ntiles + nt) * 32 + lane) << 4) + e] = f2bf(B[i]);
  }
}

// ---------------------------------------------------------------- TextCNN
// One block (8 wave32) per paper. LDS holds the [86 x 128] bf16 token tile
// (80 real + 6 zero pad rows). im2col window at position t is the contiguous
// run xs[t*128 .. t*128+h*128). 16 jobs = (4 convs) x (4 N-tiles); 2 per wave.
// All job parameters are forced into SGPRs (readfirstlane) so every loop is
// scalar-controlled and EXEC stays all-1s around the WMMAs.
__global__ __launch_bounds__(256)
void textcnn_kernel(const int* __restrict__ pf,
                    const unsigned short* __restrict__ emb,    // bf16 [50000,128]
                    const unsigned short* __restrict__ wpack,  // frag-packed
                    const float* __restrict__ cb0, const float* __restrict__ cb1,
                    const float* __restrict__ cb2, const float* __restrict__ cb3,
                    const float* __restrict__ bn_g, const float* __restrict__ bn_b,
                    const float* __restrict__ bn_m, const float* __restrict__ bn_v,
                    unsigned short* __restrict__ feats)        // bf16 [N,256]
{
  __shared__ unsigned short xs[88 * 128];
  const int p = blockIdx.x;
  unsigned* xs32 = (unsigned*)xs;
  const unsigned* emb32 = (const unsigned*)emb;

  // gather: 86 rows x 64 dwords (rows >= 80 zeroed)
  for (int idx = threadIdx.x; idx < 86 * 64; idx += 256) {
    int row = idx >> 6, col = idx & 63;
    unsigned v = 0;
    if (row < SEQ_LEN) {
      int tok = pf[p * SEQ_LEN + row];
      v = emb32[tok * 64 + col];
    }
    xs32[row * 64 + col] = v;
  }
  __syncthreads();

  const int lane = threadIdx.x & 31;
  const int g = lane >> 4;        // half-wave: K half (A) / K group (B)
  const int m = lane & 15;        // A row / B column within tile

  for (int jj = 0; jj < 2; ++jj) {
    // scalarize the per-wave job parameters
    int job = __builtin_amdgcn_readfirstlane((threadIdx.x >> 5) + jj * 8);
    int ci = job >> 2;            // conv index 0..3
    int nt = job & 3;             // filter tile 0..3
    int h  = (ci == 0) ? 2 : (ci == 1) ? 3 : (ci == 2) ? 5 : 7;
    int P  = SEQ_LEN - h + 1;     // valid positions
    int nK = h * 4;               // K steps of 32 (K = h*128)
    int wOff = (ci == 0) ? 0 : (ci == 1) ? 16384 : (ci == 2) ? 40960 : 81920;
    const float* cb = (ci == 0) ? cb0 : (ci == 1) ? cb1 : (ci == 2) ? cb2 : cb3;
    const uint4* wq = (const uint4*)(wpack + wOff);

    float mx = -3.4e38f;
    for (int mt = 0; mt < 5; ++mt) {           // 5 x 16 = 80 positions
      int t = mt * 16 + m;                     // A row for this lane
      v8f c = {};
      for (int kt = 0; kt < nK; ++kt) {
        // A fragment: 2x ds_load_b128 (16B aligned: t*64 + kt*16 + g*4 dwords)
        union { v16bf v; uint4 q[2]; } Af;
        const uint4* ap = (const uint4*)(xs32 + t * 64 + kt * 16 + g * 4);
        Af.q[0] = ap[0];
        Af.q[1] = ap[2];                       // +8 dwords = K+16 half
        // B fragment: 2x global_load_b128 from packed layout
        union { v16bf v; uint4 q[2]; } Bf;
        const uint4* bp = wq + (((kt * 4 + nt) * 32 + lane) << 1);
        Bf.q[0] = bp[0];
        Bf.q[1] = bp[1];
        c = __builtin_amdgcn_wmma_f32_16x16x32_bf16(
                false, Af.v, false, Bf.v, (short)0, c, false, false);
      }
      // accumulator row r of lane -> position mt*16 + g*8 + r
      if (mt < 4) {                            // all 16 positions valid
        #pragma unroll
        for (int r = 0; r < 8; ++r) mx = fmaxf(mx, c[r]);
      } else {                                 // mask tail positions >= P
        #pragma unroll
        for (int r = 0; r < 8; ++r) {
          int tt = 64 + g * 8 + r;
          mx = (tt < P) ? fmaxf(mx, c[r]) : mx;
        }
      }
    }
    mx = fmaxf(mx, __shfl_xor(mx, 16, 32));   // merge the two M halves
    int f  = nt * 16 + m;
    int ch = ci * NFILT + f;
    float val = fmaxf(mx + cb[f], 0.0f);                // max(relu(y+b))
    val = (val - bn_m[ch]) * rsqrtf(bn_v[ch] + 1e-5f) * bn_g[ch] + bn_b[ch];
    if (lane < 16) feats[p * TXT_DIM + ch] = f2bf(val);
  }
}

// ---------------------------------------------------------------- GEMM
// C[M,N] = A[M,K](bf16 row-major) * Bpack(frag-packed bf16), f32 accum.
// Wave = 16x64 tile, block of 8 waves = 128x64, grid = (ceil(M/128), N/64).
__global__ __launch_bounds__(256)
void gemm_bf16_kernel(const unsigned short* __restrict__ Amat,
                      const unsigned short* __restrict__ Bpack,
                      float* __restrict__ Cmat, int M, int N, int K)
{
  const int wave = threadIdx.x >> 5;
  const int lane = threadIdx.x & 31;
  const int g = lane >> 4;
  const int m = lane & 15;
  const int mbase = blockIdx.x * 128 + wave * 16;
  const int nb = blockIdx.y * 64;
  const int ntiles = N >> 4;
  const int rowc = min(mbase + m, M - 1);     // clamp for tail block
  const unsigned* A32 = (const unsigned*)Amat;
  const uint4* Bq = (const uint4*)Bpack;

  v8f c0 = {}, c1 = {}, c2 = {}, c3 = {};
  for (int kt = 0; kt < (K >> 5); ++kt) {
    __builtin_prefetch(Amat + rowc * K + kt * 32 + 64, 0, 1);
    // A fragment: 2x global_load_b128 (dword base rowc*K/2 + kt*16 + g*4)
    union { v16bf v; uint4 q[2]; } Af;
    const uint4* ap = (const uint4*)(A32 + ((rowc * K) >> 1) + kt * 16 + g * 4);
    Af.q[0] = ap[0];
    Af.q[1] = ap[2];
    // 4 B fragments, each 2x global_load_b128 from packed layout
    union { v16bf v; uint4 q[2]; } B0, B1, B2, B3;
    int tb = (kt * ntiles + (blockIdx.y << 2)) * 32 + lane;
    B0.q[0] = Bq[(tb      ) << 1]; B0.q[1] = Bq[((tb      ) << 1) + 1];
    B1.q[0] = Bq[(tb + 32 ) << 1]; B1.q[1] = Bq[((tb + 32 ) << 1) + 1];
    B2.q[0] = Bq[(tb + 64 ) << 1]; B2.q[1] = Bq[((tb + 64 ) << 1) + 1];
    B3.q[0] = Bq[(tb + 96 ) << 1]; B3.q[1] = Bq[((tb + 96 ) << 1) + 1];
    c0 = __builtin_amdgcn_wmma_f32_16x16x32_bf16(false, Af.v, false, B0.v, (short)0, c0, false, false);
    c1 = __builtin_amdgcn_wmma_f32_16x16x32_bf16(false, Af.v, false, B1.v, (short)0, c1, false, false);
    c2 = __builtin_amdgcn_wmma_f32_16x16x32_bf16(false, Af.v, false, B2.v, (short)0, c2, false, false);
    c3 = __builtin_amdgcn_wmma_f32_16x16x32_bf16(false, Af.v, false, B3.v, (short)0, c3, false, false);
  }
  #pragma unroll
  for (int r = 0; r < 8; ++r) {
    int rr = mbase + g * 8 + r;
    if (rr < M) {
      float* cp = Cmat + rr * N + nb + m;
      cp[0] = c0[r]; cp[16] = c1[r]; cp[32] = c2[r]; cp[48] = c3[r];
    }
  }
}

// ---------------------------------------------------------------- graph side
__global__ void deg_init_kernel(float* __restrict__ deg, int n) {
  int i = blockIdx.x * blockDim.x + threadIdx.x;
  if (i < n) deg[i] = 1.0f;                    // self loop
}
__global__ void deg_acc_kernel(const int* __restrict__ dst,
                               float* __restrict__ deg, int nE) {
  for (int e = blockIdx.x * blockDim.x + threadIdx.x; e < nE;
       e += gridDim.x * blockDim.x)
    atomicAdd(&deg[dst[e]], 1.0f);
}
__global__ void dinv_kernel(const float* __restrict__ deg,
                            float* __restrict__ dinv, int n) {
  int i = blockIdx.x * blockDim.x + threadIdx.x;
  if (i < n) dinv[i] = rsqrtf(deg[i]);         // deg >= 1 always
}

// scatter-add: agg[dst] += dinv[s]*dinv[d] * xw[src], 4 channels per thread
__global__ void agg_kernel(const int* __restrict__ src, const int* __restrict__ dst,
                           const float* __restrict__ dinv, const float* __restrict__ xw,
                           float* __restrict__ agg, int C4, int nE) {
  long total = (long)nE * C4;
  for (long i = (long)blockIdx.x * blockDim.x + threadIdx.x; i < total;
       i += (long)gridDim.x * blockDim.x) {
    int e  = (int)(i / C4);
    int cq = (int)(i % C4);
    int s = src[e], d = dst[e];
    float nrm = dinv[s] * dinv[d];
    float4 v = ((const float4*)xw)[(long)s * C4 + cq];
    float* ap = agg + ((long)d * C4 + cq) * 4;
    atomicAdd(ap + 0, nrm * v.x);
    atomicAdd(ap + 1, nrm * v.y);
    atomicAdd(ap + 2, nrm * v.z);
    atomicAdd(ap + 3, nrm * v.w);
  }
}

// h1 = relu(agg + dinv^2 * xw + b) -> bf16 (feeds next WMMA GEMM)
__global__ void post_relu_bf16_kernel(const float* __restrict__ agg,
                                      const float* __restrict__ xw,
                                      const float* __restrict__ dinv,
                                      const float* __restrict__ b,
                                      unsigned short* __restrict__ hout,
                                      int C, int n) {
  for (int i = blockIdx.x * blockDim.x + threadIdx.x; i < n;
       i += gridDim.x * blockDim.x) {
    int row = i / C, c = i - row * C;
    float di = dinv[row];
    float v = agg[i] + di * di * xw[i] + b[c];
    hout[i] = f2bf(fmaxf(v, 0.0f));
  }
}

// h2 = agg + dinv^2 * xw + b (f32, no relu)
__global__ void post_f32_kernel(const float* __restrict__ agg,
                                const float* __restrict__ xw,
                                const float* __restrict__ dinv,
                                const float* __restrict__ b,
                                float* __restrict__ hout, int C, int n) {
  for (int i = blockIdx.x * blockDim.x + threadIdx.x; i < n;
       i += gridDim.x * blockDim.x) {
    int row = i / C, c = i - row * C;
    float di = dinv[row];
    hout[i] = agg[i] + di * di * xw[i] + b[c];
  }
}

// L2-normalize + gather only the 5120 selected rows (one wave per row)
__global__ __launch_bounds__(256)
void norm_gather_kernel(const float* __restrict__ h2,
                        const int* __restrict__ bridge,
                        const int* __restrict__ sel,
                        float* __restrict__ out) {
  int wid  = (blockIdx.x * 256 + threadIdx.x) >> 5;
  int lane = threadIdx.x & 31;
  if (wid >= N_BRIDGE + N_SEL) return;
  int row = (wid < N_BRIDGE) ? bridge[wid] : sel[wid - N_BRIDGE];
  float4 v = ((const float4*)h2)[(long)row * 32 + lane];  // 128 f32 = 32 f4
  float ss = v.x * v.x + v.y * v.y + v.z * v.z + v.w * v.w;
  #pragma unroll
  for (int off = 16; off > 0; off >>= 1) ss += __shfl_xor(ss, off, 32);
  float sc = 1.0f / fmaxf(sqrtf(ss), 1e-12f);
  float4 o = {v.x * sc, v.y * sc, v.z * sc, v.w * sc};
  ((float4*)out)[(long)wid * 32 + lane] = o;
}

// ---------------------------------------------------------------- launcher
extern "C" void kernel_launch(void* const* d_in, const int* in_sizes, int n_in,
                              void* d_out, int out_size, void* d_ws, size_t ws_size,
                              hipStream_t stream) {
  (void)in_sizes; (void)n_in; (void)out_size; (void)ws_size;
  const int*   pf     = (const int*)d_in[0];
  const int*   edges  = (const int*)d_in[1];            // [2, nE]
  const int*   bridge = (const int*)d_in[2];
  const int*   sel    = (const int*)d_in[3];
  const float* embf   = (const float*)d_in[4];
  const float* cw[4]  = {(const float*)d_in[5], (const float*)d_in[7],
                         (const float*)d_in[9], (const float*)d_in[11]};
  const float* cb[4]  = {(const float*)d_in[6], (const float*)d_in[8],
                         (const float*)d_in[10], (const float*)d_in[12]};
  const float* bn_g = (const float*)d_in[13];
  const float* bn_b = (const float*)d_in[14];
  const float* bn_m = (const float*)d_in[15];
  const float* bn_v = (const float*)d_in[16];
  const float* w1 = (const float*)d_in[17];
  const float* b1 = (const float*)d_in[18];
  const float* w2 = (const float*)d_in[19];
  const float* b2 = (const float*)d_in[20];
  const int* esrc = edges;
  const int* edst = edges + N_EDGES;

  // workspace bump allocator (256B aligned)
  char* ws = (char*)d_ws;
  size_t off = 0;
  auto alloc = [&](size_t bytes) -> char* {
    char* p = ws + off;
    off += (bytes + 255) & ~(size_t)255;
    return p;
  };
  unsigned short* embB  = (unsigned short*)alloc((size_t)50000 * 128 * 2);
  unsigned short* wcP   = (unsigned short*)alloc((size_t)139264 * 2);
  unsigned short* w1P   = (unsigned short*)alloc((size_t)TXT_DIM * MID * 2);
  unsigned short* w2P   = (unsigned short*)alloc((size_t)MID * OUT_DIM * 2);
  unsigned short* feats = (unsigned short*)alloc((size_t)N_PAPERS * TXT_DIM * 2);
  float* xw1  = (float*)alloc((size_t)N_PAPERS * MID * 4);
  float* agg1 = (float*)alloc((size_t)N_PAPERS * MID * 4);
  unsigned short* h1 = (unsigned short*)alloc((size_t)N_PAPERS * MID * 2);
  float* xw2  = (float*)alloc((size_t)N_PAPERS * OUT_DIM * 4);
  float* agg2 = (float*)alloc((size_t)N_PAPERS * OUT_DIM * 4);
  float* h2   = (float*)alloc((size_t)N_PAPERS * OUT_DIM * 4);
  float* deg  = (float*)alloc((size_t)N_PAPERS * 4);
  float* dinv = (float*)alloc((size_t)N_PAPERS * 4);

  // 1. convert embed table to bf16; pack all weights into fragment layout
  cvt_bf16_kernel<<<4096, 256, 0, stream>>>(embf, embB, 50000 * 128);
  {
    const int hs[4]   = {2, 3, 5, 7};
    const int wOff[4] = {0, 16384, 40960, 81920};
    for (int i = 0; i < 4; ++i)
      pack_b_kernel<<<256, 256, 0, stream>>>(cw[i], wcP + wOff[i],
                                             hs[i] * EMBED, NFILT);
    pack_b_kernel<<<256, 256, 0, stream>>>(w1, w1P, TXT_DIM, MID);
    pack_b_kernel<<<128, 256, 0, stream>>>(w2, w2P, MID, OUT_DIM);
  }

  // 2. TextCNN + BN -> feats (bf16)
  textcnn_kernel<<<N_PAPERS, 256, 0, stream>>>(
      pf, embB, wcP, cb[0], cb[1], cb[2], cb[3], bn_g, bn_b, bn_m, bn_v, feats);

  // 3. degrees / norm coefficients
  deg_init_kernel<<<(N_PAPERS + 255) / 256, 256, 0, stream>>>(deg, N_PAPERS);
  deg_acc_kernel<<<2048, 256, 0, stream>>>(edst, deg, N_EDGES);
  dinv_kernel<<<(N_PAPERS + 255) / 256, 256, 0, stream>>>(deg, dinv, N_PAPERS);

  // 4. GCN layer 1
  gemm_bf16_kernel<<<dim3(118, MID / 64), 256, 0, stream>>>(
      feats, w1P, xw1, N_PAPERS, MID, TXT_DIM);
  hipMemsetAsync(agg1, 0, (size_t)N_PAPERS * MID * 4, stream);
  agg_kernel<<<4096, 256, 0, stream>>>(esrc, edst, dinv, xw1, agg1, MID / 4, N_EDGES);
  post_relu_bf16_kernel<<<4096, 256, 0, stream>>>(
      agg1, xw1, dinv, b1, h1, MID, N_PAPERS * MID);

  // 5. GCN layer 2
  gemm_bf16_kernel<<<dim3(118, OUT_DIM / 64), 256, 0, stream>>>(
      h1, w2P, xw2, N_PAPERS, OUT_DIM, MID);
  hipMemsetAsync(agg2, 0, (size_t)N_PAPERS * OUT_DIM * 4, stream);
  agg_kernel<<<4096, 256, 0, stream>>>(esrc, edst, dinv, xw2, agg2, OUT_DIM / 4, N_EDGES);
  post_f32_kernel<<<4096, 256, 0, stream>>>(
      agg2, xw2, dinv, b2, h2, OUT_DIM, N_PAPERS * OUT_DIM);

  // 6. normalize + gather only selected rows
  norm_gather_kernel<<<(N_BRIDGE + N_SEL) * 32 / 256, 256, 0, stream>>>(
      h2, bridge, sel, (float*)d_out);
}
